// FHNAttentionNoJIT_11570641896143
// MI455X (gfx1250) — matrix operand order repaired and checked by reading, not compiled, not run on hardware
//
#include <hip/hip_runtime.h>
#include <hip/hip_bf16.h>

// ---------------------------------------------------------------------------
// FHN spectral attention, MI455X (gfx1250, wave32, WMMA 16x16x32 bf16).
// v3: 2-stage unrolled software pipeline with peeled epilogue (no register
// rotation -> no accumulator copies / hazard NOPs), uniform-base + 32-bit
// lane-offset addressing (saddr+voffset loads, SALU K-advance), 32x64 wave
// tiles, LDS-transposed coalesced f32 output stores.
// ---------------------------------------------------------------------------

typedef __bf16 bf16_t;
typedef __attribute__((ext_vector_type(16))) __bf16 v16bf;
typedef __attribute__((ext_vector_type(8)))  float  v8f;

#define B_DIM  8
#define T_DIM  4096
#define D_DIM  768
#define H_DIM  12
#define HD_DIM 64
#define K_DIM  32
#define M_DIM  (B_DIM * T_DIM)   // 32768
#define N_QKV  (3 * D_DIM)       // 2304

union FragBF { v16bf v; uint4 q[2]; };
union Pack8  { bf16_t h[8]; uint4 q; };
union Pack4  { bf16_t h[4]; uint2 u; };

__device__ __forceinline__ v8f v8f_zero() {
  v8f z = {0.f, 0.f, 0.f, 0.f, 0.f, 0.f, 0.f, 0.f};
  return z;
}

// A-matrix 16x32 bf16 fragment: lane holds row M = lane%16.
// lanes 0-15: K = {0..7, 16..23}; lanes 16-31: K = {8..15, 24..31}.
// base = uniform pointer (advanced by k-step); off = per-lane element offset
// that already folds in the (lane&16 ? 8 : 0) K-half adjustment.
__device__ __forceinline__ v16bf load_a_frag(const bf16_t* __restrict__ base, int off) {
  FragBF f;
  f.q[0] = *(const uint4*)(base + off);
  f.q[1] = *(const uint4*)(base + off + 16);
  return f.v;
}

// B-matrix 32x16 bf16 fragment: lane holds column N = lane%16.
// off folds in the (lane&16 ? 8 : 0) K-half adjustment.
__device__ __forceinline__ v16bf load_b_frag(const bf16_t* __restrict__ base, int off) {
  FragBF f;
  f.q[0] = *(const uint4*)(base + off);
  f.q[1] = *(const uint4*)(base + off + 16);
  return f.v;
}

__device__ __forceinline__ v8f wmma_bf16(v16bf a, v16bf b, v8f c) {
  return __builtin_amdgcn_wmma_f32_16x16x32_bf16(false, a, false, b,
                                                 (short)0, c, false, false);
}

// ---------------------------------------------------------------------------
// Conversion kernels
// ---------------------------------------------------------------------------
__global__ void k_f32_to_bf16_x4(const float* __restrict__ in,
                                 bf16_t* __restrict__ out, int n4) {
  int i = blockIdx.x * blockDim.x + threadIdx.x;
  if (i >= n4) return;
  float4 f = ((const float4*)in)[i];
  Pack4 p;
  p.h[0] = (bf16_t)f.x; p.h[1] = (bf16_t)f.y;
  p.h[2] = (bf16_t)f.z; p.h[3] = (bf16_t)f.w;
  ((uint2*)out)[i] = p.u;
}

// basis -> bf16 [B,T,K] and transposed bf16 [B,K,T]
__global__ void k_basis_conv(const float* __restrict__ in,
                             bf16_t* __restrict__ basb,
                             bf16_t* __restrict__ basT) {
  int i = blockIdx.x * blockDim.x + threadIdx.x;  // over B*T*K
  int b   = i / (T_DIM * K_DIM);
  int rem = i - b * (T_DIM * K_DIM);
  int t = rem / K_DIM;
  int k = rem - t * K_DIM;
  bf16_t v = (bf16_t)in[i];
  basb[i] = v;
  basT[((size_t)b * K_DIM + k) * T_DIM + t] = v;
}

// ---------------------------------------------------------------------------
// Kernel 1: qkv = x @ w_qkv^T  (M=32768, N=2304, K=768), store transposed
// per-head bf16: qkvT[3][B][H][HD][T].
// Block = 8 waves as 4(M) x 2(N); wave tile 32(M) x 64(N) = 8 accumulators.
// 2-stage pipeline, step 64 (two 32-wide K-steps in flight). Grid (256, 18).
// ---------------------------------------------------------------------------
__global__ void __launch_bounds__(256) k_qkv_gemm(const bf16_t* __restrict__ xb,
                                                  const bf16_t* __restrict__ wb,
                                                  bf16_t* __restrict__ qkvT) {
  const int lane = threadIdx.x & 31;
  const int l15  = lane & 15;
  const int wave = threadIdx.x >> 5;
  const int mw = wave >> 1, nw = wave & 1;
  const int m0 = blockIdx.x * 128 + mw * 32;
  const int n0 = blockIdx.y * 128 + nw * 64;

  const int aadj = (lane & 16) ? 8 : 0;
  const int badj = (lane & 16) ? 8 : 0;
  const int ao0 = (m0 + l15) * D_DIM + aadj;
  const int ao1 = ao0 + 16 * D_DIM;
  const int bo0 = (n0 +  0 + l15) * D_DIM + badj;
  const int bo1 = bo0 + 16 * D_DIM;
  const int bo2 = bo0 + 32 * D_DIM;
  const int bo3 = bo0 + 48 * D_DIM;

  v8f acc[2][4];
#pragma unroll
  for (int mi = 0; mi < 2; ++mi)
#pragma unroll
    for (int j = 0; j < 4; ++j) acc[mi][j] = v8f_zero();

  // stage 0 @ k=0
  v16bf sA0 = load_a_frag(xb, ao0);
  v16bf sA1 = load_a_frag(xb, ao1);
  v16bf sB0 = load_b_frag(wb, bo0);
  v16bf sB1 = load_b_frag(wb, bo1);
  v16bf sB2 = load_b_frag(wb, bo2);
  v16bf sB3 = load_b_frag(wb, bo3);

  for (int k0 = 0; k0 < D_DIM - 64; k0 += 64) {
    // stage 1 @ k0+32
    v16bf tA0 = load_a_frag(xb + k0 + 32, ao0);
    v16bf tA1 = load_a_frag(xb + k0 + 32, ao1);
    v16bf tB0 = load_b_frag(wb + k0 + 32, bo0);
    v16bf tB1 = load_b_frag(wb + k0 + 32, bo1);
    v16bf tB2 = load_b_frag(wb + k0 + 32, bo2);
    v16bf tB3 = load_b_frag(wb + k0 + 32, bo3);
    acc[0][0] = wmma_bf16(sA0, sB0, acc[0][0]);
    acc[1][0] = wmma_bf16(sA1, sB0, acc[1][0]);
    acc[0][1] = wmma_bf16(sA0, sB1, acc[0][1]);
    acc[1][1] = wmma_bf16(sA1, sB1, acc[1][1]);
    acc[0][2] = wmma_bf16(sA0, sB2, acc[0][2]);
    acc[1][2] = wmma_bf16(sA1, sB2, acc[1][2]);
    acc[0][3] = wmma_bf16(sA0, sB3, acc[0][3]);
    acc[1][3] = wmma_bf16(sA1, sB3, acc[1][3]);
    // stage 0 @ k0+64 (fresh defs, no rotation copies)
    sA0 = load_a_frag(xb + k0 + 64, ao0);
    sA1 = load_a_frag(xb + k0 + 64, ao1);
    sB0 = load_b_frag(wb + k0 + 64, bo0);
    sB1 = load_b_frag(wb + k0 + 64, bo1);
    sB2 = load_b_frag(wb + k0 + 64, bo2);
    sB3 = load_b_frag(wb + k0 + 64, bo3);
    acc[0][0] = wmma_bf16(tA0, tB0, acc[0][0]);
    acc[1][0] = wmma_bf16(tA1, tB0, acc[1][0]);
    acc[0][1] = wmma_bf16(tA0, tB1, acc[0][1]);
    acc[1][1] = wmma_bf16(tA1, tB1, acc[1][1]);
    acc[0][2] = wmma_bf16(tA0, tB2, acc[0][2]);
    acc[1][2] = wmma_bf16(tA1, tB2, acc[1][2]);
    acc[0][3] = wmma_bf16(tA0, tB3, acc[0][3]);
    acc[1][3] = wmma_bf16(tA1, tB3, acc[1][3]);
  }
  {
    // epilogue: stage0 holds k = D-64; load final k = D-32
    v16bf tA0 = load_a_frag(xb + D_DIM - 32, ao0);
    v16bf tA1 = load_a_frag(xb + D_DIM - 32, ao1);
    v16bf tB0 = load_b_frag(wb + D_DIM - 32, bo0);
    v16bf tB1 = load_b_frag(wb + D_DIM - 32, bo1);
    v16bf tB2 = load_b_frag(wb + D_DIM - 32, bo2);
    v16bf tB3 = load_b_frag(wb + D_DIM - 32, bo3);
    acc[0][0] = wmma_bf16(sA0, sB0, acc[0][0]);
    acc[1][0] = wmma_bf16(sA1, sB0, acc[1][0]);
    acc[0][1] = wmma_bf16(sA0, sB1, acc[0][1]);
    acc[1][1] = wmma_bf16(sA1, sB1, acc[1][1]);
    acc[0][2] = wmma_bf16(sA0, sB2, acc[0][2]);
    acc[1][2] = wmma_bf16(sA1, sB2, acc[1][2]);
    acc[0][3] = wmma_bf16(sA0, sB3, acc[0][3]);
    acc[1][3] = wmma_bf16(sA1, sB3, acc[1][3]);
    acc[0][0] = wmma_bf16(tA0, tB0, acc[0][0]);
    acc[1][0] = wmma_bf16(tA1, tB0, acc[1][0]);
    acc[0][1] = wmma_bf16(tA0, tB1, acc[0][1]);
    acc[1][1] = wmma_bf16(tA1, tB1, acc[1][1]);
    acc[0][2] = wmma_bf16(tA0, tB2, acc[0][2]);
    acc[1][2] = wmma_bf16(tA1, tB2, acc[1][2]);
    acc[0][3] = wmma_bf16(tA0, tB3, acc[0][3]);
    acc[1][3] = wmma_bf16(tA1, tB3, acc[1][3]);
  }

  // D layout: lane holds column n; VGPR r holds row m=r (+8 for upper lanes).
  const int bidx = m0 >> 12;
#pragma unroll
  for (int mi = 0; mi < 2; ++mi) {
    const int tbase = ((m0 + mi * 16) & 4095) + ((lane & 16) ? 8 : 0);
#pragma unroll
    for (int j = 0; j < 4; ++j) {
      int n     = n0 + j * 16 + l15;
      int three = n / D_DIM;
      int rem   = n - three * D_DIM;
      int h     = rem >> 6;
      int hd    = rem & 63;
      size_t base =
          ((((size_t)(three * B_DIM + bidx)) * H_DIM + h) * HD_DIM + hd) * T_DIM + tbase;
      Pack8 p;
#pragma unroll
      for (int r = 0; r < 8; ++r) p.h[r] = (bf16_t)acc[mi][j][r];
      *(uint4*)(qkvT + base) = p.q;   // 8 bf16, t-contiguous -> b128 store
    }
  }
}

// ---------------------------------------------------------------------------
// Kernel 2: spec[which,b,h,d,k] = sum_t qkvT[which,b,h,d,t] * basisT[b,k,t]
// One block per (which,b,h); 8 waves = 4 d-tiles x 2 k-tiles.
// Dual accumulators, 2 K-steps in flight, loads issued right after consume.
// ---------------------------------------------------------------------------
__global__ void __launch_bounds__(256) k_spec_gemm(const bf16_t* __restrict__ qkvT,
                                                   const bf16_t* __restrict__ basT,
                                                   float* __restrict__ spec) {
  const int gid   = blockIdx.x;                 // which*96 + b*12 + h
  const int which = gid / (B_DIM * H_DIM);
  const int rem   = gid - which * (B_DIM * H_DIM);
  const int b = rem / H_DIM, h = rem - (rem / H_DIM) * H_DIM;
  const int lane = threadIdx.x & 31;
  const int l15  = lane & 15;
  const int wave = threadIdx.x >> 5;
  const int mtile = wave >> 1;                  // 0..3 (d)
  const int ntile = wave & 1;                   // 0..1 (k)
  const int d    = mtile * 16 + l15;
  const int kcol = ntile * 16 + l15;

  const bf16_t* abase =
      qkvT + ((((size_t)which * B_DIM + b) * H_DIM + h) * HD_DIM) * T_DIM;
  const bf16_t* bbase = basT + (size_t)b * K_DIM * T_DIM;
  const int aoff = d * T_DIM + ((lane & 16) ? 8 : 0);
  const int boff = kcol * T_DIM + ((lane & 16) ? 8 : 0);

  v8f acc0 = v8f_zero(), acc1 = v8f_zero();
  v16bf a0 = load_a_frag(abase, aoff);
  v16bf c0 = load_b_frag(bbase, boff);
  v16bf a1 = load_a_frag(abase + 32, aoff);
  v16bf c1 = load_b_frag(bbase + 32, boff);
  for (int t0 = 0; t0 < T_DIM - 64; t0 += 64) {
    acc0 = wmma_bf16(a0, c0, acc0);
    a0 = load_a_frag(abase + t0 + 64, aoff);
    c0 = load_b_frag(bbase + t0 + 64, boff);
    acc1 = wmma_bf16(a1, c1, acc1);
    a1 = load_a_frag(abase + t0 + 96, aoff);
    c1 = load_b_frag(bbase + t0 + 96, boff);
  }
  acc0 = wmma_bf16(a0, c0, acc0);
  acc1 = wmma_bf16(a1, c1, acc1);

  float* out = spec + (((size_t)which * B_DIM + b) * H_DIM + h) * (HD_DIM * K_DIM);
  const int mbase = mtile * 16 + ((lane & 16) ? 8 : 0);
#pragma unroll
  for (int r = 0; r < 8; ++r)
    out[(size_t)(mbase + r) * K_DIM + kcol] = acc0[r] + acc1[r];
}

// ---------------------------------------------------------------------------
// Kernel 3: attn dot over d, filter sigmoid, FHN (2 unrolled steps), fuse
// fhn * v_spec -> bf16 osT[b,h,d,k].  grid = B*H, 32 lanes = k index.
// ---------------------------------------------------------------------------
__global__ void k_fhn(const float* __restrict__ spec,
                      const float* __restrict__ sfilt,
                      bf16_t* __restrict__ osT) {
  const int bh = blockIdx.x;           // b*H + h
  const int h  = bh % H_DIM;
  const int k  = threadIdx.x;          // 0..31
  const size_t qoff = ((size_t)0 * B_DIM * H_DIM + bh) * (HD_DIM * K_DIM);
  const size_t koff = ((size_t)1 * B_DIM * H_DIM + bh) * (HD_DIM * K_DIM);
  const size_t voff = ((size_t)2 * B_DIM * H_DIM + bh) * (HD_DIM * K_DIM);

  float attn = 0.f;
  for (int d = 0; d < HD_DIM; ++d)
    attn += spec[qoff + d * K_DIM + k] * spec[koff + d * K_DIM + k];
  attn *= 0.125f;  // 1/sqrt(64)

  float filt = 1.f / (1.f + __expf(-sfilt[h * 32 + k]));
  float s = attn * filt;

  float as    = fabsf(s);
  float scale = fmaxf(as, 1e-6f);
  float sn    = s / scale;
  float gate  = 1.f / (1.f + __expf(-(as - 0.5f) * 10.f));
  float I     = sn * (0.1f + 0.9f * gate);
  const float alpha = 1.0f / 12.5f;          // DT/TAU
  const float denom = 1.0f + alpha * 0.8f;   // 1 + alpha*B_PARAM
  float v = 0.f, w = 0.f;
#pragma unroll
  for (int it = 0; it < 2; ++it) {
    float dv = v - (v * v * v) * (1.f / 3.f) - w + I;
    float vn = v + dv;                       // DT = 1
    float wn = (w + (vn + 0.7f) * alpha) / denom;
    v = fminf(fmaxf(vn, -3.f), 3.f);
    w = fminf(fmaxf(wn, -3.f), 3.f);
  }
  float r = v * scale;

  bf16_t* o = osT + (size_t)bh * (HD_DIM * K_DIM);
  for (int d = 0; d < HD_DIM; ++d)
    o[d * K_DIM + k] = (bf16_t)(r * spec[voff + d * K_DIM + k]);
}

// ---------------------------------------------------------------------------
// Kernel 4: y[b,t,h*64+d] = sum_k osT[b,h,d,k] * basis[b,t,k]
// C[d,t] = osT(64x32) x basis-view(32xT): single WMMA K-step.
// All 4 B fragments hoisted before the WMMAs. grid (B*H, 32).
// ---------------------------------------------------------------------------
__global__ void __launch_bounds__(256) k_recon_gemm(const bf16_t* __restrict__ osT,
                                                    const bf16_t* __restrict__ basb,
                                                    bf16_t* __restrict__ yb) {
  const int bh = blockIdx.x;
  const int b = bh / H_DIM, h = bh - (bh / H_DIM) * H_DIM;
  const int lane = threadIdx.x & 31;
  const int l15  = lane & 15;
  const int wave = threadIdx.x >> 5;
  const int mtile = wave & 3;                          // d tile 0..3
  const int t0    = (blockIdx.y * 2 + (wave >> 2)) * 64;
  const int d     = mtile * 16 + l15;

  const bf16_t* abase = osT + (size_t)bh * (HD_DIM * K_DIM);
  const bf16_t* bbase = basb + (size_t)b * T_DIM * K_DIM;
  v16bf a = load_a_frag(abase, d * K_DIM + ((lane & 16) ? 8 : 0));
  const int badj = (lane & 16) ? 8 : 0;
  v16bf bb[4];
#pragma unroll
  for (int j = 0; j < 4; ++j)
    bb[j] = load_b_frag(bbase, (t0 + j * 16 + l15) * K_DIM + badj);

  const int mbase = mtile * 16 + ((lane & 16) ? 8 : 0);
#pragma unroll
  for (int j = 0; j < 4; ++j) {
    int t = t0 + j * 16 + l15;                         // D column = time index
    v8f acc = wmma_bf16(a, bb[j], v8f_zero());
    Pack8 p;
#pragma unroll
    for (int r = 0; r < 8; ++r) p.h[r] = (bf16_t)acc[r];
    size_t base = ((size_t)b * T_DIM + t) * D_DIM + h * HD_DIM + mbase;
    *(uint4*)(yb + base) = p.q;                        // 8 bf16, d-contiguous
  }
}

// ---------------------------------------------------------------------------
// Kernel 5: out = y @ w_out^T  (M=32768, N=768, K=768), f32 output.
// Same pipelined 32x64 wave tile as kernel 1; f32 results staged through a
// wave-private padded LDS buffer (16x68, two passes) for coalesced b128
// global stores. Grid (256, 6).
// ---------------------------------------------------------------------------
__global__ void __launch_bounds__(256) k_out_gemm(const bf16_t* __restrict__ yb,
                                                  const bf16_t* __restrict__ wob,
                                                  float* __restrict__ out) {
  __shared__ float xpose[8][16][68];   // 34816 B; per-wave 16x64 tile, padded
  const int lane = threadIdx.x & 31;
  const int l15  = lane & 15;
  const int wave = threadIdx.x >> 5;
  const int mw = wave >> 1, nw = wave & 1;
  const int m0 = blockIdx.x * 128 + mw * 32;
  const int n0 = blockIdx.y * 128 + nw * 64;

  const int ao0 = (m0 + l15) * D_DIM + ((lane & 16) ? 8 : 0);
  const int ao1 = ao0 + 16 * D_DIM;
  const int bo0 = (n0 + l15) * D_DIM + ((lane & 16) ? 8 : 0);
  const int bo1 = bo0 + 16 * D_DIM;
  const int bo2 = bo0 + 32 * D_DIM;
  const int bo3 = bo0 + 48 * D_DIM;

  v8f acc[2][4];
#pragma unroll
  for (int mi = 0; mi < 2; ++mi)
#pragma unroll
    for (int j = 0; j < 4; ++j) acc[mi][j] = v8f_zero();

  v16bf sA0 = load_a_frag(yb, ao0);
  v16bf sA1 = load_a_frag(yb, ao1);
  v16bf sB0 = load_b_frag(wob, bo0);
  v16bf sB1 = load_b_frag(wob, bo1);
  v16bf sB2 = load_b_frag(wob, bo2);
  v16bf sB3 = load_b_frag(wob, bo3);

  for (int k0 = 0; k0 < D_DIM - 64; k0 += 64) {
    v16bf tA0 = load_a_frag(yb + k0 + 32, ao0);
    v16bf tA1 = load_a_frag(yb + k0 + 32, ao1);
    v16bf tB0 = load_b_frag(wob + k0 + 32, bo0);
    v16bf tB1 = load_b_frag(wob + k0 + 32, bo1);
    v16bf tB2 = load_b_frag(wob + k0 + 32, bo2);
    v16bf tB3 = load_b_frag(wob + k0 + 32, bo3);
    acc[0][0] = wmma_bf16(sA0, sB0, acc[0][0]);
    acc[1][0] = wmma_bf16(sA1, sB0, acc[1][0]);
    acc[0][1] = wmma_bf16(sA0, sB1, acc[0][1]);
    acc[1][1] = wmma_bf16(sA1, sB1, acc[1][1]);
    acc[0][2] = wmma_bf16(sA0, sB2, acc[0][2]);
    acc[1][2] = wmma_bf16(sA1, sB2, acc[1][2]);
    acc[0][3] = wmma_bf16(sA0, sB3, acc[0][3]);
    acc[1][3] = wmma_bf16(sA1, sB3, acc[1][3]);
    sA0 = load_a_frag(yb + k0 + 64, ao0);
    sA1 = load_a_frag(yb + k0 + 64, ao1);
    sB0 = load_b_frag(wob + k0 + 64, bo0);
    sB1 = load_b_frag(wob + k0 + 64, bo1);
    sB2 = load_b_frag(wob + k0 + 64, bo2);
    sB3 = load_b_frag(wob + k0 + 64, bo3);
    acc[0][0] = wmma_bf16(tA0, tB0, acc[0][0]);
    acc[1][0] = wmma_bf16(tA1, tB0, acc[1][0]);
    acc[0][1] = wmma_bf16(tA0, tB1, acc[0][1]);
    acc[1][1] = wmma_bf16(tA1, tB1, acc[1][1]);
    acc[0][2] = wmma_bf16(tA0, tB2, acc[0][2]);
    acc[1][2] = wmma_bf16(tA1, tB2, acc[1][2]);
    acc[0][3] = wmma_bf16(tA0, tB3, acc[0][3]);
    acc[1][3] = wmma_bf16(tA1, tB3, acc[1][3]);
  }
  {
    v16bf tA0 = load_a_frag(yb + D_DIM - 32, ao0);
    v16bf tA1 = load_a_frag(yb + D_DIM - 32, ao1);
    v16bf tB0 = load_b_frag(wob + D_DIM - 32, bo0);
    v16bf tB1 = load_b_frag(wob + D_DIM - 32, bo1);
    v16bf tB2 = load_b_frag(wob + D_DIM - 32, bo2);
    v16bf tB3 = load_b_frag(wob + D_DIM - 32, bo3);
    acc[0][0] = wmma_bf16(sA0, sB0, acc[0][0]);
    acc[1][0] = wmma_bf16(sA1, sB0, acc[1][0]);
    acc[0][1] = wmma_bf16(sA0, sB1, acc[0][1]);
    acc[1][1] = wmma_bf16(sA1, sB1, acc[1][1]);
    acc[0][2] = wmma_bf16(sA0, sB2, acc[0][2]);
    acc[1][2] = wmma_bf16(sA1, sB2, acc[1][2]);
    acc[0][3] = wmma_bf16(sA0, sB3, acc[0][3]);
    acc[1][3] = wmma_bf16(sA1, sB3, acc[1][3]);
    acc[0][0] = wmma_bf16(tA0, tB0, acc[0][0]);
    acc[1][0] = wmma_bf16(tA1, tB0, acc[1][0]);
    acc[0][1] = wmma_bf16(tA0, tB1, acc[0][1]);
    acc[1][1] = wmma_bf16(tA1, tB1, acc[1][1]);
    acc[0][2] = wmma_bf16(tA0, tB2, acc[0][2]);
    acc[1][2] = wmma_bf16(tA1, tB2, acc[1][2]);
    acc[0][3] = wmma_bf16(tA0, tB3, acc[0][3]);
    acc[1][3] = wmma_bf16(tA1, tB3, acc[1][3]);
  }

  // Two passes of wave-private LDS transpose -> coalesced b128 global stores.
  float (*tb)[68] = xpose[wave];
  const int mloc  = (lane & 16) ? 8 : 0;
  const int rr    = l15;
  const int cbase = (lane & 16) ? 32 : 0;
#pragma unroll
  for (int mi = 0; mi < 2; ++mi) {
#pragma unroll
    for (int j = 0; j < 4; ++j)
#pragma unroll
      for (int r = 0; r < 8; ++r)
        tb[mloc + r][j * 16 + l15] = acc[mi][j][r];
    // same-wave DS ops are in-order; compiler inserts s_wait_dscnt for RAW
    float* orow = out + (size_t)(m0 + mi * 16 + rr) * D_DIM + n0 + cbase;
#pragma unroll
    for (int q = 0; q < 8; ++q)
      *(float4*)(orow + q * 4) = *(const float4*)&tb[rr][cbase + q * 4];
  }
}

// ---------------------------------------------------------------------------
extern "C" void kernel_launch(void* const* d_in, const int* in_sizes, int n_in,
                              void* d_out, int out_size, void* d_ws, size_t ws_size,
                              hipStream_t stream) {
  (void)in_sizes; (void)n_in; (void)out_size; (void)ws_size;
  const float* x     = (const float*)d_in[0];
  const float* basis = (const float*)d_in[1];
  const float* wqkv  = (const float*)d_in[2];
  const float* wout  = (const float*)d_in[3];
  const float* sfilt = (const float*)d_in[4];
  float* out = (float*)d_out;
  char* ws = (char*)d_ws;

  size_t off = 0;
  bf16_t* xb   = (bf16_t*)(ws + off); off += (size_t)M_DIM * D_DIM * 2;
  bf16_t* wqb  = (bf16_t*)(ws + off); off += (size_t)N_QKV * D_DIM * 2;
  bf16_t* wob  = (bf16_t*)(ws + off); off += (size_t)D_DIM * D_DIM * 2;
  bf16_t* basb = (bf16_t*)(ws + off); off += (size_t)B_DIM * T_DIM * K_DIM * 2;
  bf16_t* basT = (bf16_t*)(ws + off); off += (size_t)B_DIM * T_DIM * K_DIM * 2;
  bf16_t* qkvT = (bf16_t*)(ws + off); off += (size_t)3 * B_DIM * H_DIM * HD_DIM * T_DIM * 2;
  float*  spec = (float*)(ws + off);  off += (size_t)3 * B_DIM * H_DIM * HD_DIM * K_DIM * 4;
  bf16_t* osT  = (bf16_t*)(ws + off); off += (size_t)B_DIM * H_DIM * HD_DIM * K_DIM * 2;
  bf16_t* yb   = (bf16_t*)(ws + off);

  // 0) precision conversion (bf16 copies live in the 192MB L2)
  {
    int n4 = M_DIM * D_DIM / 4;
    k_f32_to_bf16_x4<<<(n4 + 255) / 256, 256, 0, stream>>>(x, xb, n4);
  }
  {
    int n4 = N_QKV * D_DIM / 4;
    k_f32_to_bf16_x4<<<(n4 + 255) / 256, 256, 0, stream>>>(wqkv, wqb, n4);
  }
  {
    int n4 = D_DIM * D_DIM / 4;
    k_f32_to_bf16_x4<<<(n4 + 255) / 256, 256, 0, stream>>>(wout, wob, n4);
  }
  {
    int n = B_DIM * T_DIM * K_DIM;
    k_basis_conv<<<(n + 255) / 256, 256, 0, stream>>>(basis, basb, basT);
  }

  // 1) QKV projection (116 GFLOP)
  k_qkv_gemm<<<dim3(M_DIM / 128, N_QKV / 128), 256, 0, stream>>>(xb, wqb, qkvT);
  // 2) spectral projections of q,k,v (4.8 GFLOP)
  k_spec_gemm<<<3 * B_DIM * H_DIM, 256, 0, stream>>>(qkvT, basT, spec);
  // 3) attention dot + FHN dynamics + fhn*v_spec (tiny)
  k_fhn<<<B_DIM * H_DIM, 32, 0, stream>>>(spec, sfilt, osT);
  // 4) spectral -> time reconstruction (1.6 GFLOP)
  k_recon_gemm<<<dim3(B_DIM * H_DIM, T_DIM / 128), 256, 0, stream>>>(osT, basb, yb);
  // 5) output projection (38.7 GFLOP)
  k_out_gemm<<<dim3(M_DIM / 128, D_DIM / 128), 256, 0, stream>>>(yb, wob, out);
}
